// AttentionAugmentedSDE_73959336837175
// MI455X (gfx1250) — compile-verified, matrix operand order checked by reading, not run on hardware
//
#include <hip/hip_runtime.h>
#include <hip/hip_bf16.h>
#include <stdint.h>
#include <math.h>

// ---------------------------------------------------------------------------
// AttentionAugmentedSDE for MI455X (gfx1250, wave32).
//   - all GEMMs: f16-in / f32-accumulate WMMA 16x16x32
//   - every LDS tile is a straight async DMA copy (GLOBAL_LOAD_ASYNC_TO_LDS_B128,
//     ASYNCcnt); B matrices pre-transposed to [N,K]
//   - double-buffered LDS: next tile streams in during WMMA on current tile,
//     synchronized with partial s_wait_asynccnt
//   - SDE scan: ONE kernel per step (2 WGs: drift+diffusion MLPs fused with the
//     Euler-Maruyama update; inter-WG rendezvous via atomic + cluster barrier)
//   - f16 weights (~34 MB transposed) stay resident in the 192 MB L2
// ---------------------------------------------------------------------------

typedef _Float16 f16;
typedef __attribute__((ext_vector_type(16))) _Float16 v16h;
typedef __attribute__((ext_vector_type(8)))  float    v8f;

constexpr int Bsz = 64, Ssz = 1024, Dsz = 512, Hsz = 512, Lsz = 1024, NLsz = 4;

#define BM 64
#define BN 128
#define BK 32
#define LDT 40  // padded LDS stride (BK + 8) in halves

// Async global->LDS 16B copy (per-lane). Passing the real LDS pointer through
// the asm (as ptrtoint) both captures the shared array -- so the "memory"
// clobber is known to alias it and the later ds_loads are NOT folded away --
// and supplies the architecturally-correct VDST value (low 32 bits of a
// generic LDS address are the LDS byte offset).
__device__ __forceinline__ void async_g2l_b128(void* lds, const void* gaddr) {
  asm volatile("global_load_async_to_lds_b128 %0, %1, off"
               :: "v"((uint32_t)(uintptr_t)lds), "v"((uint64_t)(uintptr_t)gaddr)
               : "memory");
}
// Wait until at most N async ops remain outstanding (in-order completion).
template <int N>
__device__ __forceinline__ void wait_async() {
  asm volatile("s_wait_asynccnt %0" :: "n"(N) : "memory");
}

// Build a 16x32 (16-bit) WMMA operand fragment from an LDS tile whose 32
// K-values per row are contiguous (stride `ldh` halves). CDNA5 ISA 7.12.2.
__device__ __forceinline__ v16h ld_frag(const f16* base, int row, int ldh, int g) {
  union { v16h v; uint32_t u[8]; } f;
  const uint32_t* p = (const uint32_t*)(base + (size_t)row * ldh);
#pragma unroll
  for (int i = 0; i < 4; ++i) {
    f.u[i]     = p[i + 4 * g];
    f.u[i + 4] = p[8 + i + 4 * g];
  }
  return f.v;
}

// ---------------------------------------------------------------------------
// Tiled WMMA GEMM:  C = act(scale * A @ Bt^T + bias)
//   A:  [M,K] f16 (lda==K)      Bt: [N,K] f16 (row = output column)
//   C: f32 (Cf) and/or f16 (Ch), leading dim ldc; batched via blockIdx.z.
// BM=64 x BN=128 x BK=32; 256 threads = 8 waves, wave tile 32x32.
// Double-buffered LDS, all staging via async DMA (3 ops/wave/tile).
// ---------------------------------------------------------------------------
__global__ __launch_bounds__(256) void gemm_f16_wmma(
    const f16* __restrict__ A, const f16* __restrict__ Bt,
    const float* __restrict__ bias, float* __restrict__ Cf, f16* __restrict__ Ch,
    int M, int N, int K, int ldc, int act, float scale,
    long strideA, long strideB, long strideC) {
  constexpr int TILE_H = BM * LDT + BN * LDT;  // halves per buffer
  __shared__ f16 tile[2 * TILE_H];

  const int tid  = threadIdx.x;
  const int wid  = tid >> 5;
  const int lane = tid & 31;
  const int g    = lane >> 4;
  const int r    = lane & 15;
  const int wm   = wid >> 2;  // 0..1
  const int wn   = wid & 3;   // 0..3
  const int m0   = blockIdx.y * BM;
  const int n0   = blockIdx.x * BN;

  const size_t bz = blockIdx.z;
  A  += bz * (size_t)strideA;
  Bt += bz * (size_t)strideB;
  if (Cf) Cf += bz * (size_t)strideC;
  if (Ch) Ch += bz * (size_t)strideC;

  v8f acc[2][2] = {};
  const int T = K / BK;

  auto stage = [&](int kt, int bsel) {
    f16* buf = tile + bsel * TILE_H;
    const int k0 = kt * BK;
    {  // A tile [BM][BK]
      const int i  = tid >> 2;
      const int jc = (tid & 3) * 8;
      async_g2l_b128(buf + i * LDT + jc, A + (size_t)(m0 + i) * K + k0 + jc);
    }
#pragma unroll
    for (int c = 0; c < 2; ++c) {  // Bt tile [BN][BK]
      const int cc = tid + c * 256;
      const int n  = cc >> 2;
      const int kc = (cc & 3) * 8;
      async_g2l_b128(buf + BM * LDT + n * LDT + kc,
                     Bt + (size_t)(n0 + n) * K + k0 + kc);
    }
  };

  stage(0, 0);
  for (int kt = 0; kt < T; ++kt) {
    const int cur = kt & 1;
    if (kt + 1 < T) {
      __syncthreads();            // everyone done reading buffer cur^1
      stage(kt + 1, cur ^ 1);     // stream next tile during compute
      wait_async<3>();            // current tile (3 older ops) has landed
    } else {
      wait_async<0>();
    }
    __syncthreads();

    const f16* Alds = tile + cur * TILE_H;
    const f16* Blds = Alds + BM * LDT;
    v16h af[2], bf[2];
    af[0] = ld_frag(Alds, wm * 32 + r, LDT, g);
    af[1] = ld_frag(Alds, wm * 32 + 16 + r, LDT, g);
    bf[0] = ld_frag(Blds, wn * 32 + r, LDT, g);
    bf[1] = ld_frag(Blds, wn * 32 + 16 + r, LDT, g);
#pragma unroll
    for (int ms = 0; ms < 2; ++ms)
#pragma unroll
      for (int ns = 0; ns < 2; ++ns)
        acc[ms][ns] = __builtin_amdgcn_wmma_f32_16x16x32_f16(
            false, af[ms], false, bf[ns], (short)0, acc[ms][ns], false, false);
  }

#pragma unroll
  for (int ms = 0; ms < 2; ++ms)
#pragma unroll
    for (int ns = 0; ns < 2; ++ns) {
      const int col  = n0 + wn * 32 + ns * 16 + r;
      const float bc = bias ? bias[col] : 0.0f;
#pragma unroll
      for (int j = 0; j < 8; ++j) {
        const int row = m0 + wm * 32 + ms * 16 + j + 8 * g;
        float x = acc[ms][ns][j] * scale + bc;
        if (act == 1)      x = fmaxf(x, 0.0f);
        else if (act == 2) x = 1.0f / (1.0f + __expf(-x));
        if (Cf) Cf[(size_t)row * ldc + col] = x;
        if (Ch) Ch[(size_t)row * ldc + col] = (f16)x;
      }
    }
  (void)M;
}

// ---------------------------------------------------------------------------
// Fused SDE step kernel: grid = 2 WGs (0=drift, 1=diffusion), 512 threads.
// ---------------------------------------------------------------------------
struct SdeStepArgs {
  const f16 *h16;
  const f16 *drin, *drblk, *drout;      // drift weights, [N,K] f16
  const f16 *diin, *diblk, *diout;      // diffusion weights, [N,K] f16
  const float *drinB, *drblkB, *droutB;
  const float *diinB, *diblkB, *dioutB;
  f16 *actA0, *actA1, *actB0, *actB1;   // per-WG activation ping-pong [64,1024]
  float *fF, *gF, *hF;
  f16 *h16w;
  const float *noise;                   // this step's dW [64,512]
  f16 *concat;
  int *sync;
  int step, expect;
  float dt, sdt;
};

constexpr int SDE_AHALVES  = 64 * LDT;                   // A tile halves
constexpr int SDE_BUFH     = 64 * LDT + 16 * 64 * LDT;   // halves per buffer
constexpr int SDE_BUFBYTES = SDE_BUFH * 2;

// One layer: out = act(Ain[64,K] @ WT^T + bias), WT: [N,K], N = 256*NSUB.
// Double-buffered; A tile async; wave-private WT slice async (2*NSUB ops/lane).
template <int NSUB>
__device__ void mlp_layer(const f16* __restrict__ Ain, int K,
                          const f16* __restrict__ WT, const float* __restrict__ bias,
                          int act, f16* __restrict__ out16, float* __restrict__ outF,
                          f16* smem) {
  const int N    = 256 * NSUB;
  const int tid  = threadIdx.x;
  const int wid  = tid >> 5;
  const int lane = tid & 31;
  const int g    = lane >> 4;
  const int r    = lane & 15;
  const int n0   = wid * NSUB * 16;

  v8f acc[4][NSUB] = {};
  const int T = K / BK;

  auto stage = [&](int kt, int bsel) {
    f16* buf = smem + bsel * SDE_BUFH;
    const int k0 = kt * BK;
    if (tid < 256) {  // A tile [64][32]
      const int i  = tid >> 2;
      const int jc = (tid & 3) * 8;
      async_g2l_b128(buf + i * LDT + jc, Ain + (size_t)i * K + k0 + jc);
    }
#pragma unroll
    for (int c = 0; c < 2 * NSUB; ++c) {  // WT slice [NSUB*16][32], wave-private
      const int cc = lane + c * 32;
      const int n  = cc >> 2;
      const int kc = (cc & 3) * 8;
      async_g2l_b128(buf + SDE_AHALVES + (wid * 64 + n) * LDT + kc,
                     WT + (size_t)(n0 + n) * K + k0 + kc);
    }
  };

  stage(0, 0);
  for (int kt = 0; kt < T; ++kt) {
    const int cur = kt & 1;
    if (kt + 1 < T) {
      __syncthreads();
      stage(kt + 1, cur ^ 1);
      wait_async<2 * NSUB>();  // >= ops of current tile still outstanding
    } else {
      wait_async<0>();
    }
    __syncthreads();

    const f16* As = smem + cur * SDE_BUFH;
    const f16* Bs = As + SDE_AHALVES + wid * 64 * LDT;
    v16h af[4];
#pragma unroll
    for (int m = 0; m < 4; ++m) af[m] = ld_frag(As, m * 16 + r, LDT, g);
#pragma unroll
    for (int ns = 0; ns < NSUB; ++ns) {
      const v16h bf = ld_frag(Bs, ns * 16 + r, LDT, g);
#pragma unroll
      for (int m = 0; m < 4; ++m)
        acc[m][ns] = __builtin_amdgcn_wmma_f32_16x16x32_f16(
            false, af[m], false, bf, (short)0, acc[m][ns], false, false);
    }
  }

#pragma unroll
  for (int ns = 0; ns < NSUB; ++ns) {
    const int col  = n0 + ns * 16 + r;
    const float bc = bias[col];
#pragma unroll
    for (int m = 0; m < 4; ++m)
#pragma unroll
      for (int j = 0; j < 8; ++j) {
        const int row = m * 16 + j + 8 * g;
        float x = acc[m][ns][j] + bc;
        if (act == 1)      x = fmaxf(x, 0.0f);
        else if (act == 2) x = 1.0f / (1.0f + __expf(-x));
        if (out16) out16[(size_t)row * N + col] = (f16)x;
        if (outF)  outF[(size_t)row * N + col] = x;
      }
  }
}

__global__ __launch_bounds__(512) void sde_step(SdeStepArgs a) {
  extern __shared__ f16 smem[];
  const int tid = threadIdx.x;
  const int wg  = blockIdx.x;  // 0 = drift, 1 = diffusion

  const f16 *inW  = wg ? a.diin  : a.drin;
  const f16 *bkW  = wg ? a.diblk : a.drblk;
  const f16 *otW  = wg ? a.diout : a.drout;
  const float *inB = wg ? a.diinB  : a.drinB;
  const float *bkB = wg ? a.diblkB : a.drblkB;
  const float *otB = wg ? a.dioutB : a.droutB;
  f16 *p0 = wg ? a.actB0 : a.actA0;
  f16 *p1 = wg ? a.actB1 : a.actA1;
  float *outF = wg ? a.gF : a.fF;

  mlp_layer<4>(a.h16, Hsz, inW, inB, 0, p0, nullptr, smem);
  __threadfence(); __syncthreads();
  f16 *cur = p0, *nxt = p1;
  for (int i = 0; i < NLsz; ++i) {
    mlp_layer<4>(cur, Lsz, bkW + (size_t)i * Lsz * Lsz, bkB + (size_t)i * Lsz,
                 1, nxt, nullptr, smem);
    __threadfence(); __syncthreads();
    f16* t = cur; cur = nxt; nxt = t;
  }
  mlp_layer<2>(cur, Lsz, otW, otB, 2, nullptr, outF, smem);
  __threadfence(); __syncthreads();

  // rendezvous between the two WGs
  if (tid == 0) atomicAdd(a.sync, 1);
  __builtin_amdgcn_s_cluster_barrier();  // CDNA5 cluster split-barrier (NOP unclustered)
  if (tid == 0) {
    while (atomicAdd(a.sync, 0) < a.expect) __builtin_amdgcn_s_sleep(2);
  }
  __syncthreads();
  __threadfence();

  // Euler-Maruyama update: each WG handles half of [64,512]
  const int half = (Bsz * Hsz) / 2;
  for (int i = tid; i < half; i += 512) {
    const int idx = wg * half + i;
    const int b = idx / Hsz, j = idx % Hsz;
    const float hn = a.hF[idx] + a.fF[idx] * a.dt + a.gF[idx] * a.sdt * a.noise[idx];
    a.hF[idx]   = hn;
    a.h16w[idx] = (f16)hn;
    a.concat[((size_t)b * Ssz + (a.step + 1)) * (Dsz + Hsz) + Dsz + j] = (f16)hn;
  }
}

// ---------------------------------------------------------------------------
// Row softmax: f32 scores -> f16 probabilities.
// ---------------------------------------------------------------------------
__global__ __launch_bounds__(256) void softmax_rows(
    const float* __restrict__ Sc, f16* __restrict__ P, int n) {
  __shared__ float red[256];
  const size_t row = blockIdx.x;
  const float* src = Sc + row * (size_t)n;
  f16* dst = P + row * (size_t)n;

  float mx = -3.0e38f;
  for (int j = threadIdx.x; j < n; j += 256) mx = fmaxf(mx, src[j]);
  red[threadIdx.x] = mx;
  __syncthreads();
  for (int s = 128; s > 0; s >>= 1) {
    if (threadIdx.x < s) red[threadIdx.x] = fmaxf(red[threadIdx.x], red[threadIdx.x + s]);
    __syncthreads();
  }
  mx = red[0];
  __syncthreads();
  float sum = 0.0f;
  for (int j = threadIdx.x; j < n; j += 256) sum += __expf(src[j] - mx);
  red[threadIdx.x] = sum;
  __syncthreads();
  for (int s = 128; s > 0; s >>= 1) {
    if (threadIdx.x < s) red[threadIdx.x] += red[threadIdx.x + s];
    __syncthreads();
  }
  const float inv = 1.0f / red[0];
  for (int j = threadIdx.x; j < n; j += 256) dst[j] = (f16)(__expf(src[j] - mx) * inv);
}

// ---------------------------------------------------------------------------
// Elementwise / transpose helpers.
// ---------------------------------------------------------------------------
__global__ void conv_f32_to_f16(const float* __restrict__ in, f16* __restrict__ out, long n) {
  long i = (long)blockIdx.x * blockDim.x + threadIdx.x;
  const long stride = (long)gridDim.x * blockDim.x;
  for (; i < n; i += stride) out[i] = (f16)in[i];
}

// in: nb slices of [R,C] f32  ->  out: nb slices of [C,R] f16 (write-coalesced)
__global__ void convT_f32_to_f16(const float* __restrict__ in, f16* __restrict__ out,
                                 int R, int C, long total) {
  long i = (long)blockIdx.x * blockDim.x + threadIdx.x;
  const long stride = (long)gridDim.x * blockDim.x;
  const long rc = (long)R * C;
  for (; i < total; i += stride) {
    const long s = i / rc, rem = i % rc;
    const long r = rem % R, c = rem / R;
    out[s * rc + c * R + r] = (f16)in[s * rc + r * C + c];
  }
}

// in: nb slices of [R,C] f16 -> out: nb slices of [C,R] f16
__global__ void trans_f16(const f16* __restrict__ in, f16* __restrict__ out,
                          int R, int C, long total) {
  long i = (long)blockIdx.x * blockDim.x + threadIdx.x;
  const long stride = (long)gridDim.x * blockDim.x;
  const long rc = (long)R * C;
  for (; i < total; i += stride) {
    const long s = i / rc, rem = i % rc;
    const long r = rem % R, c = rem / R;
    out[s * rc + c * R + r] = in[s * rc + r * C + c];
  }
}

__global__ void init_state(float* __restrict__ h, f16* __restrict__ h16,
                           f16* __restrict__ concat, int* __restrict__ sync) {
  const int i = blockIdx.x * blockDim.x + threadIdx.x;
  if (i == 0) *sync = 0;
  if (i >= Bsz * Hsz) return;
  const int b = i / Hsz, j = i % Hsz;
  h[i]   = 0.0f;
  h16[i] = (f16)0.0f;
  concat[((size_t)b * Ssz) * (Dsz + Hsz) + Dsz + j] = (f16)0.0f;
}

// ---------------------------------------------------------------------------
// Host driver.
// ---------------------------------------------------------------------------
extern "C" void kernel_launch(void* const* d_in, const int* in_sizes, int n_in,
                              void* d_out, int out_size, void* d_ws, size_t ws_size,
                              hipStream_t stream) {
  (void)in_sizes; (void)n_in; (void)out_size; (void)ws_size;

  const float* x       = (const float*)d_in[0];
  const float* noise   = (const float*)d_in[1];
  const float* Wq      = (const float*)d_in[2];
  const float* bq      = (const float*)d_in[3];
  const float* Wk      = (const float*)d_in[4];
  const float* bk      = (const float*)d_in[5];
  const float* Wv      = (const float*)d_in[6];
  const float* bv      = (const float*)d_in[7];
  const float* fin_W   = (const float*)d_in[8];
  const float* fin_b   = (const float*)d_in[9];
  const float* blk_W   = (const float*)d_in[10];
  const float* blk_b   = (const float*)d_in[11];
  const float* fout_W  = (const float*)d_in[12];
  const float* fout_b  = (const float*)d_in[13];
  const float* dr_in_W = (const float*)d_in[14];
  const float* dr_in_b = (const float*)d_in[15];
  const float* dr_blk_W= (const float*)d_in[16];
  const float* dr_blk_b= (const float*)d_in[17];
  const float* dr_out_W= (const float*)d_in[18];
  const float* dr_out_b= (const float*)d_in[19];
  const float* di_in_W = (const float*)d_in[20];
  const float* di_in_b = (const float*)d_in[21];
  const float* di_blk_W= (const float*)d_in[22];
  const float* di_blk_b= (const float*)d_in[23];
  const float* di_out_W= (const float*)d_in[24];
  const float* di_out_b= (const float*)d_in[25];
  float* out = (float*)d_out;

  // ---- workspace layout (256B aligned slabs) ----
  char* base = (char*)d_ws;
  size_t off = 0;
  auto a16 = [&](size_t n) { f16* p = (f16*)(base + off);
                             off += ((n * sizeof(f16)) + 255) & ~(size_t)255; return p; };
  auto a32 = [&](size_t n) { float* p = (float*)(base + off);
                             off += ((n * sizeof(float)) + 255) & ~(size_t)255; return p; };

  f16* wqT     = a16((size_t)Dsz * Dsz);      // all weights stored [N,K] f16
  f16* wkT     = a16((size_t)Dsz * Dsz);
  f16* wvT     = a16((size_t)Dsz * Dsz);
  f16* finT    = a16((size_t)(Dsz + Hsz) * Lsz);
  f16* blkT    = a16((size_t)NLsz * Lsz * Lsz);
  f16* foutT   = a16((size_t)Lsz * Dsz);
  f16* drinT   = a16((size_t)Hsz * Lsz);
  f16* drblkT  = a16((size_t)NLsz * Lsz * Lsz);
  f16* droutT  = a16((size_t)Lsz * Hsz);
  f16* diinT   = a16((size_t)Hsz * Lsz);
  f16* diblkT  = a16((size_t)NLsz * Lsz * Lsz);
  f16* dioutT  = a16((size_t)Lsz * Hsz);
  f16* x16     = a16((size_t)Bsz * Ssz * Dsz);
  f16* q16     = a16((size_t)Bsz * Ssz * Dsz);
  f16* k16     = a16((size_t)Bsz * Ssz * Dsz);
  f16* v16     = a16((size_t)Bsz * Ssz * Dsz);
  f16* vT16    = a16((size_t)Bsz * Dsz * Ssz);  // per-batch V^T for context GEMM
  float* scF   = a32((size_t)Bsz * Ssz * Ssz);
  f16* p16     = a16((size_t)Bsz * Ssz * Ssz);
  f16* concat16= a16((size_t)Bsz * Ssz * (Dsz + Hsz));
  f16* deca    = a16((size_t)Bsz * Ssz * Lsz);
  f16* decb    = a16((size_t)Bsz * Ssz * Lsz);
  float* hF    = a32((size_t)Bsz * Hsz);
  f16* h16     = a16((size_t)Bsz * Hsz);
  f16* actA0   = a16((size_t)Bsz * Lsz);
  f16* actA1   = a16((size_t)Bsz * Lsz);
  f16* actB0   = a16((size_t)Bsz * Lsz);
  f16* actB1   = a16((size_t)Bsz * Lsz);
  float* fF    = a32((size_t)Bsz * Hsz);
  float* gF    = a32((size_t)Bsz * Hsz);
  int* syncCtr = (int*)a32(64);

  auto convT = [&](const float* src, f16* dst, int R, int C, int nb) {
    convT_f32_to_f16<<<4096, 256, 0, stream>>>(src, dst, R, C, (long)nb * R * C);
  };
  auto gemm = [&](const f16* A, const f16* Bt, const float* bias, float* Cf, f16* Ch,
                  int M, int N, int K, int ldc, int act, float scale,
                  int nb, long sA, long sB, long sC) {
    dim3 grid(N / BN, M / BM, nb);
    gemm_f16_wmma<<<grid, 256, 0, stream>>>(A, Bt, bias, Cf, Ch, M, N, K, ldc,
                                            act, scale, sA, sB, sC);
  };

  // ---- precision cast + transpose: every B operand becomes [N,K] f16 ----
  conv_f32_to_f16<<<2048, 256, 0, stream>>>(x, x16, (long)Bsz * Ssz * Dsz);
  convT(Wq, wqT, Dsz, Dsz, 1);
  convT(Wk, wkT, Dsz, Dsz, 1);
  convT(Wv, wvT, Dsz, Dsz, 1);
  convT(fin_W, finT, Dsz + Hsz, Lsz, 1);
  convT(blk_W, blkT, Lsz, Lsz, NLsz);
  convT(fout_W, foutT, Lsz, Dsz, 1);
  convT(dr_in_W, drinT, Hsz, Lsz, 1);
  convT(dr_blk_W, drblkT, Lsz, Lsz, NLsz);
  convT(dr_out_W, droutT, Lsz, Hsz, 1);
  convT(di_in_W, diinT, Hsz, Lsz, 1);
  convT(di_blk_W, diblkT, Lsz, Lsz, NLsz);
  convT(di_out_W, dioutT, Lsz, Hsz, 1);

  // ---- self-attention (batched WMMA GEMMs + softmax) ----
  const int MT = Bsz * Ssz;  // 65536 token rows
  gemm(x16, wqT, bq, nullptr, q16, MT, Dsz, Dsz, Dsz, 0, 1.0f, 1, 0, 0, 0);
  gemm(x16, wkT, bk, nullptr, k16, MT, Dsz, Dsz, Dsz, 0, 1.0f, 1, 0, 0, 0);
  gemm(x16, wvT, bv, nullptr, v16, MT, Dsz, Dsz, Dsz, 0, 1.0f, 1, 0, 0, 0);
  trans_f16<<<4096, 256, 0, stream>>>(v16, vT16, Ssz, Dsz, (long)Bsz * Ssz * Dsz);

  const float iscale = 0.04419417382415922f;  // 1/sqrt(512)
  // scores = Q @ K^T : K is [S,D] token-major == [N,K] layout already
  gemm(q16, k16, nullptr, scF, nullptr, Ssz, Ssz, Dsz, Ssz, 0, iscale,
       Bsz, (long)Ssz * Dsz, (long)Ssz * Dsz, (long)Ssz * Ssz);
  softmax_rows<<<Bsz * Ssz, 256, 0, stream>>>(scF, p16, Ssz);
  // context = P @ V : Bt = V^T [D,S]; write into left half of concat
  gemm(p16, vT16, nullptr, nullptr, concat16, Ssz, Dsz, Ssz, Dsz + Hsz, 0, 1.0f,
       Bsz, (long)Ssz * Ssz, (long)Dsz * Ssz, (long)Ssz * (Dsz + Hsz));

  // ---- Euler-Maruyama SDE scan: one fused kernel per step ----
  init_state<<<(Bsz * Hsz + 255) / 256, 256, 0, stream>>>(hF, h16, concat16, syncCtr);
  const float dt  = 1.0f / (float)(Ssz - 1);
  const float sdt = sqrtf(dt);
  const size_t smemBytes = 2 * (size_t)SDE_BUFBYTES;  // 174 KB (double-buffered)

  SdeStepArgs sargs;
  sargs.h16  = h16;   sargs.h16w = h16;
  sargs.drin = drinT; sargs.drblk = drblkT; sargs.drout = droutT;
  sargs.diin = diinT; sargs.diblk = diblkT; sargs.diout = dioutT;
  sargs.drinB = dr_in_b; sargs.drblkB = dr_blk_b; sargs.droutB = dr_out_b;
  sargs.diinB = di_in_b; sargs.diblkB = di_blk_b; sargs.dioutB = di_out_b;
  sargs.actA0 = actA0; sargs.actA1 = actA1; sargs.actB0 = actB0; sargs.actB1 = actB1;
  sargs.fF = fF; sargs.gF = gF; sargs.hF = hF;
  sargs.concat = concat16; sargs.sync = syncCtr;
  sargs.dt = dt; sargs.sdt = sdt;
  for (int step = 0; step < Ssz - 1; ++step) {
    sargs.noise  = noise + (size_t)step * Bsz * Hsz;
    sargs.step   = step;
    sargs.expect = 2 * (step + 1);
    sde_step<<<dim3(2), dim3(512), smemBytes, stream>>>(sargs);
  }

  // ---- decoder MLP over all B*S tokens ----
  gemm(concat16, finT, fin_b, nullptr, deca, MT, Lsz, Dsz + Hsz, Lsz, 0, 1.0f, 1, 0, 0, 0);
  f16* cur = deca; f16* nxt = decb;
  for (int i = 0; i < NLsz; ++i) {
    gemm(cur, blkT + (size_t)i * Lsz * Lsz, blk_b + (size_t)i * Lsz,
         nullptr, nxt, MT, Lsz, Lsz, Lsz, 1, 1.0f, 1, 0, 0, 0);
    f16* t = cur; cur = nxt; nxt = t;
  }
  gemm(cur, foutT, fout_b, out, nullptr, MT, Dsz, Lsz, Dsz, 0, 1.0f, 1, 0, 0, 0);
}
